// DistilBertCRF_56315611185510
// MI455X (gfx1250) — matrix operand order repaired and checked by reading, not compiled, not run on hardware
//
#include <hip/hip_runtime.h>
#include <math.h>

// ---------------- problem constants ----------------
constexpr int B  = 32;
constexpr int S  = 512;
constexpr int H  = 768;
constexpr int L  = 6;
constexpr int NH = 12;
constexpr int FF = 3072;
constexpr int T  = 7;
constexpr int DH = 64;
constexpr int M  = B * S;          // 16384 token rows

// ---------------- WMMA types ----------------
typedef __attribute__((ext_vector_type(16))) __bf16 v16bf;
typedef __attribute__((ext_vector_type(8)))  float  v8f;

union FragBF {
    v16bf v;
    uint4 u[2];
};

static __device__ __forceinline__ v8f wmma_bf16(v16bf a, v16bf b, v8f c) {
    // D = A(16x32) * B(32x16) + C, f32 accumulate
    return __builtin_amdgcn_wmma_f32_16x16x32_bf16(
        /*neg_a=*/false, a, /*neg_b=*/false, b,
        /*c_mod=*/(short)0, c, /*reuse_a=*/false, /*reuse_b=*/false);
}

// Async copy of 16 bytes global -> LDS (gfx1250 GLOBAL_LOAD_ASYNC_TO_LDS_B128,
// tracked on ASYNCcnt). lds_addr is the raw 32-bit LDS byte address
// (ISA 10.2: flat LDS pointer low 32 bits == LDS offset).
static __device__ __forceinline__ void async_copy_b128(unsigned lds_addr,
                                                       unsigned long long gaddr) {
    asm volatile("global_load_async_to_lds_b128 %0, %1, off"
                 :: "v"(lds_addr), "v"(gaddr) : "memory");
}
static __device__ __forceinline__ void wait_asynccnt0() {
    asm volatile("s_wait_asynccnt 0" ::: "memory");
}

// ---------------- block reduction helper (256 threads, 8 waves) ----------------
static __device__ __forceinline__ float block_sum_256(float v, float* red) {
#pragma unroll
    for (int off = 16; off > 0; off >>= 1) v += __shfl_xor(v, off, 32);
    const int wid = threadIdx.x >> 5;
    if ((threadIdx.x & 31) == 0) red[wid] = v;
    __syncthreads();
    float s = 0.f;
#pragma unroll
    for (int w = 0; w < 8; ++w) s += red[w];
    __syncthreads();
    return s;
}

// =====================================================================
// Weight pre-pass: f32 [K,N] row-major  ->  bf16 [N,K] row-major.
// Runs once per launch over all 36 weight matrices (~85 MB out, ~10 us
// at 23.3 TB/s) so the GEMM hot loop can stage B tiles as raw async
// byte copies with no transpose and half the bytes.
// grid = (N/64, K/64), 256 threads.
// =====================================================================
__global__ __launch_bounds__(256)
void wconv_kernel(const float* __restrict__ in, __bf16* __restrict__ out,
                  int Kdim, int Ndim) {
    __shared__ __bf16 tile[64][72];
    const int n0 = blockIdx.x * 64;
    const int k0 = blockIdx.y * 64;
    for (int i = threadIdx.x; i < 64 * 64; i += 256) {
        const int r = i >> 6, c = i & 63;          // r = k, c = n (coalesced on c)
        tile[c][r] = (__bf16)in[(size_t)(k0 + r) * Ndim + n0 + c];
    }
    __syncthreads();
    for (int i = threadIdx.x; i < 64 * 64; i += 256) {
        const int r = i >> 6, c = i & 63;          // r = n, c = k (coalesced on c)
        out[(size_t)(n0 + r) * Kdim + k0 + c] = tile[r][c];
    }
}

// =====================================================================
// Embedding + LayerNorm:  x[row] = LN(word_emb[id] + pos_emb[s])
// =====================================================================
__global__ __launch_bounds__(256)
void embed_ln_kernel(const int* __restrict__ ids,
                     const float* __restrict__ wemb,
                     const float* __restrict__ pemb,
                     const float* __restrict__ gam,
                     const float* __restrict__ bet,
                     float* __restrict__ X) {
    __shared__ float red[8];
    const int row = blockIdx.x;
    const int s   = row & (S - 1);
    const int id  = ids[row];
    float vals[3];
    float sum = 0.f;
#pragma unroll
    for (int i = 0; i < 3; ++i) {
        const int c = threadIdx.x + 256 * i;
        const float v = wemb[(size_t)id * H + c] + pemb[(size_t)s * H + c];
        vals[i] = v;
        sum += v;
    }
    const float mean = block_sum_256(sum, red) * (1.0f / H);
    float sq = 0.f;
#pragma unroll
    for (int i = 0; i < 3; ++i) {
        const float d = vals[i] - mean;
        sq += d * d;
    }
    const float var  = block_sum_256(sq, red) * (1.0f / H);
    const float rstd = rsqrtf(var + 1e-12f);
#pragma unroll
    for (int i = 0; i < 3; ++i) {
        const int c = threadIdx.x + 256 * i;
        X[(size_t)row * H + c] = (vals[i] - mean) * rstd * gam[c] + bet[c];
    }
}

// =====================================================================
// Residual add + LayerNorm:  x[row] = LN(x[row] + y[row]) (in place on x)
// =====================================================================
__global__ __launch_bounds__(256)
void add_ln_kernel(float* __restrict__ X,
                   const float* __restrict__ Y,
                   const float* __restrict__ gam,
                   const float* __restrict__ bet) {
    __shared__ float red[8];
    const int row = blockIdx.x;
    float vals[3];
    float sum = 0.f;
#pragma unroll
    for (int i = 0; i < 3; ++i) {
        const int c = threadIdx.x + 256 * i;
        const float v = X[(size_t)row * H + c] + Y[(size_t)row * H + c];
        vals[i] = v;
        sum += v;
    }
    const float mean = block_sum_256(sum, red) * (1.0f / H);
    float sq = 0.f;
#pragma unroll
    for (int i = 0; i < 3; ++i) {
        const float d = vals[i] - mean;
        sq += d * d;
    }
    const float var  = block_sum_256(sq, red) * (1.0f / H);
    const float rstd = rsqrtf(var + 1e-12f);
#pragma unroll
    for (int i = 0; i < 3; ++i) {
        const int c = threadIdx.x + 256 * i;
        X[(size_t)row * H + c] = (vals[i] - mean) * rstd * gam[c] + bet[c];
    }
}

// =====================================================================
// GEMM  C = act(A[M,K](f32) x Bt[N,K](bf16)^T + bias)  via bf16 WMMA.
// Block: 256 threads = 8 waves; tile 128x128x32.
// A tile: f32 -> bf16 VALU conversion into LDS (overlaps async B copy).
// B tile: raw async byte copy from pre-transposed bf16 weights (ASYNCcnt).
// Requires M%128==0, N%128==0, K%32==0 (true for all shapes used here).
// =====================================================================
template <bool GELU>
__global__ __launch_bounds__(256)
void gemm_bias_kernel(const float* __restrict__ A,
                      const __bf16* __restrict__ Bt,   // [N][K] bf16
                      const float* __restrict__ bias,
                      float* __restrict__ C,
                      int Mdim, int Ndim, int Kdim) {
    (void)Mdim;
    __shared__ __align__(16) __bf16 As[128 * 40];   // [row][k], stride 40
    __shared__ __align__(16) __bf16 Bs[128 * 40];   // [col][k], stride 40

    const int t    = threadIdx.x;
    const int wid  = t >> 5;
    const int lane = t & 31;
    const int half = lane >> 4;
    const int l15  = lane & 15;

    const int m0 = blockIdx.y * 128;
    const int n0 = blockIdx.x * 128;
    const int wm = (wid & 3) * 32;   // wave row offset in tile
    const int wn = (wid >> 2) * 64;  // wave col offset in tile

    const int abase = half ? 8 : 0;   // A-frag K sub-offset per lane half
    const int bbase = half ? 8 : 0;   // B-frag K sub-offset per lane half

    v8f acc[2][4] = {};

    for (int k0 = 0; k0 < Kdim; k0 += 32) {
        __syncthreads();

        // ---- issue async B-tile copy: 128 rows x 64B, 512 x b128 chunks ----
#pragma unroll
        for (int i = 0; i < 2; ++i) {
            const int idx = t + 256 * i;          // chunk id 0..511
            const int row = idx >> 2;             // 4 chunks per row
            const int ch  = (idx & 3) * 16;       // byte offset in row
            const unsigned long long ga =
                (unsigned long long)(const void*)(Bt + (size_t)(n0 + row) * Kdim + k0) + ch;
            const unsigned la =
                (unsigned)(unsigned long long)(const void*)&Bs[row * 40] + ch;
            async_copy_b128(la, ga);
        }

        // ---- stage A tile (128x32 f32 -> bf16), coalesced float4 loads ----
#pragma unroll
        for (int i = 0; i < 4; ++i) {
            const int idx = t + 256 * i;          // float4 index, 0..1023
            const int row = idx >> 3;             // 8 float4 per row
            const int c4  = (idx & 7) * 4;
            const float4 v = *(const float4*)(A + (size_t)(m0 + row) * Kdim + k0 + c4);
            __bf16* d = &As[row * 40 + c4];
            d[0] = (__bf16)v.x; d[1] = (__bf16)v.y;
            d[2] = (__bf16)v.z; d[3] = (__bf16)v.w;
            if (k0 + 32 < Kdim)
                __builtin_prefetch(A + (size_t)(m0 + row) * Kdim + k0 + 32 + c4, 0, 1);
        }

        wait_asynccnt0();
        __syncthreads();

        // ---- fragment loads (contiguous 16B ds reads matching ISA layout) ----
        FragBF a[2], bf[4];
#pragma unroll
        for (int mt = 0; mt < 2; ++mt) {
            const int row = wm + mt * 16 + l15;
            a[mt].u[0] = *(const uint4*)&As[row * 40 + abase];       // K 0..7 (+8)
            a[mt].u[1] = *(const uint4*)&As[row * 40 + 16 + abase];  // K 16..23 (+8)
        }
#pragma unroll
        for (int nt = 0; nt < 4; ++nt) {
            const int col = wn + nt * 16 + l15;
            bf[nt].u[0] = *(const uint4*)&Bs[col * 40 + bbase];      // K 0..7 (+8)
            bf[nt].u[1] = *(const uint4*)&Bs[col * 40 + bbase + 16]; // K 16..23 (+8)
        }
#pragma unroll
        for (int mt = 0; mt < 2; ++mt)
#pragma unroll
            for (int nt = 0; nt < 4; ++nt)
                acc[mt][nt] = wmma_bf16(a[mt].v, bf[nt].v, acc[mt][nt]);
    }

    // ---- epilogue: bias (+ exact GELU), f32 store ----
#pragma unroll
    for (int mt = 0; mt < 2; ++mt)
#pragma unroll
        for (int nt = 0; nt < 4; ++nt)
#pragma unroll
            for (int r = 0; r < 8; ++r) {
                const int row = m0 + wm + mt * 16 + r + half * 8;
                const int col = n0 + wn + nt * 16 + l15;
                float v = acc[mt][nt][r] + bias[col];
                if (GELU) v = 0.5f * v * (1.0f + erff(v * 0.70710678118f));
                C[(size_t)row * Ndim + col] = v;
            }
}

// =====================================================================
// Flash-style attention, one (b, h, 64-q-row) tile per block.
// 128 threads = 4 waves; each wave owns 16 query rows.
// Streams keys 32 at a time; scores and P*V via WMMA; online softmax.
// =====================================================================
__global__ __launch_bounds__(128)
void attention_kernel(const float* __restrict__ Q,
                      const float* __restrict__ Kb,
                      const float* __restrict__ Vb,
                      const int* __restrict__ amask,
                      float* __restrict__ ctx) {
    const int qb   = blockIdx.x;   // q-row tile (64 rows)
    const int h    = blockIdx.y;
    const int b    = blockIdx.z;
    const int t    = threadIdx.x;
    const int wid  = t >> 5;
    const int lane = t & 31;
    const int half = lane >> 4;
    const int l15  = lane & 15;

    __shared__ __align__(16) __bf16 Qs[64 * 72];     // [qrow][dh]
    __shared__ __align__(16) __bf16 Ks[32 * 72];     // [key][dh]
    __shared__ __align__(16) __bf16 Vt[64 * 40];     // [dh][key] (transposed)
    __shared__ __align__(16) __bf16 Ps[4][16 * 40];  // per-wave P scratch [row][key]
    __shared__ float maskc[32];

    const size_t rowQ0 = (size_t)b * S + qb * 64;
    const int hcol = h * DH;

    // stage Q tile (64x64), f32 -> bf16
    for (int i = t; i < 64 * 64; i += 128) {
        const int r = i >> 6, c = i & 63;
        Qs[r * 72 + c] = (__bf16)Q[(rowQ0 + r) * H + hcol + c];
    }
    __syncthreads();

    const int abase = half ? 8 : 0;
    const int kb2   = half ? 8 : 0;
    const int qr    = wid * 16 + l15;

    FragBF a_q[2];
    a_q[0].u[0] = *(const uint4*)&Qs[qr * 72 + abase];
    a_q[0].u[1] = *(const uint4*)&Qs[qr * 72 + 16 + abase];
    a_q[1].u[0] = *(const uint4*)&Qs[qr * 72 + 32 + abase];
    a_q[1].u[1] = *(const uint4*)&Qs[qr * 72 + 48 + abase];

    v8f o_acc[4] = {};
    float m_r[8], l_r[8];
#pragma unroll
    for (int r = 0; r < 8; ++r) { m_r[r] = -1e30f; l_r[r] = 0.f; }

    for (int j = 0; j < S; j += 32) {
        __syncthreads();
        // stage K (row-major) and V (transposed), bf16
        for (int i = t; i < 32 * 64; i += 128) {
            const int r = i >> 6, c = i & 63;
            const size_t g = ((size_t)b * S + j + r) * H + hcol + c;
            Ks[r * 72 + c] = (__bf16)Kb[g];
            Vt[c * 40 + r] = (__bf16)Vb[g];
        }
        if (t < 32) maskc[t] = amask[b * S + j + t] ? 0.f : -1e30f;
        __syncthreads();

        // scores: two 16-key tiles, each = Q(16x64) x K^T(64x16)
        v8f sc[2];
#pragma unroll
        for (int c = 0; c < 2; ++c) {
            const int key = c * 16 + l15;
            FragBF bk0, bk1;
            bk0.u[0] = *(const uint4*)&Ks[key * 72 + kb2];
            bk0.u[1] = *(const uint4*)&Ks[key * 72 + kb2 + 16];
            bk1.u[0] = *(const uint4*)&Ks[key * 72 + 32 + kb2];
            bk1.u[1] = *(const uint4*)&Ks[key * 72 + 32 + kb2 + 16];
            v8f s = {};
            s = wmma_bf16(a_q[0].v, bk0.v, s);
            s = wmma_bf16(a_q[1].v, bk1.v, s);
            const float madd = maskc[c * 16 + l15];
#pragma unroll
            for (int r = 0; r < 8; ++r) s[r] = s[r] * 0.125f + madd;
            sc[c] = s;
        }

        // online softmax: row max / rescale / exp / row sum
        float mnew[8], esc[8];
#pragma unroll
        for (int r = 0; r < 8; ++r) {
            float mx = fmaxf(sc[0][r], sc[1][r]);
#pragma unroll
            for (int off = 1; off < 16; off <<= 1)
                mx = fmaxf(mx, __shfl_xor(mx, off, 32));
            mnew[r] = fmaxf(m_r[r], mx);
            esc[r]  = expf(m_r[r] - mnew[r]);
        }
#pragma unroll
        for (int c = 0; c < 2; ++c)
#pragma unroll
            for (int r = 0; r < 8; ++r) {
                const float p = expf(sc[c][r] - mnew[r]);
                sc[c][r] = p;
                Ps[wid][(r + half * 8) * 40 + c * 16 + l15] = (__bf16)p;
            }
#pragma unroll
        for (int r = 0; r < 8; ++r) {
            float ps = sc[0][r] + sc[1][r];
#pragma unroll
            for (int off = 1; off < 16; off <<= 1) ps += __shfl_xor(ps, off, 32);
            l_r[r] = l_r[r] * esc[r] + ps;
            m_r[r] = mnew[r];
        }
#pragma unroll
        for (int nt = 0; nt < 4; ++nt)
#pragma unroll
            for (int r = 0; r < 8; ++r) o_acc[nt][r] *= esc[r];

        // wave-local LDS bounce: C-layout P -> A-fragment layout
        asm volatile("s_wait_dscnt 0" ::: "memory");
        __builtin_amdgcn_wave_barrier();

        FragBF pa;
        pa.u[0] = *(const uint4*)&Ps[wid][l15 * 40 + abase];
        pa.u[1] = *(const uint4*)&Ps[wid][l15 * 40 + 16 + abase];
#pragma unroll
        for (int nt = 0; nt < 4; ++nt) {
            FragBF vb;
            const int d = nt * 16 + l15;
            vb.u[0] = *(const uint4*)&Vt[d * 40 + kb2];
            vb.u[1] = *(const uint4*)&Vt[d * 40 + kb2 + 16];
            o_acc[nt] = wmma_bf16(pa.v, vb.v, o_acc[nt]);
        }
    }

    // epilogue: O / l  -> ctx[b, s, h*64 + d]
#pragma unroll
    for (int nt = 0; nt < 4; ++nt)
#pragma unroll
        for (int r = 0; r < 8; ++r) {
            const int row = qb * 64 + wid * 16 + r + half * 8;
            const int col = hcol + nt * 16 + l15;
            ctx[((size_t)b * S + row) * H + col] = o_acc[nt][r] / l_r[r];
        }
}

// =====================================================================
// Emissions: [M,768] x [768,7] + b  — 7 waves per row, one tag each.
// =====================================================================
__global__ __launch_bounds__(224)
void emissions_kernel(const float* __restrict__ X,
                      const float* __restrict__ W,
                      const float* __restrict__ bias,
                      float* __restrict__ E) {
    const int row  = blockIdx.x;
    const int tag  = threadIdx.x >> 5;   // 0..6
    const int lane = threadIdx.x & 31;
    float acc = 0.f;
    for (int c = lane; c < H; c += 32)
        acc += X[(size_t)row * H + c] * W[(size_t)c * T + tag];
#pragma unroll
    for (int off = 16; off > 0; off >>= 1) acc += __shfl_xor(acc, off, 32);
    if (lane == 0) E[(size_t)row * T + tag] = acc + bias[tag];
}

// =====================================================================
// CRF forward: one wave per batch element; lanes 0..6 hold alpha[tag].
// =====================================================================
__global__ __launch_bounds__(32)
void crf_kernel(const float* __restrict__ E,
                const int* __restrict__ labels,
                const float* __restrict__ start,
                const float* __restrict__ endw,
                const float* __restrict__ trans,
                float* __restrict__ lossb) {
    const int b    = blockIdx.x;
    const int lane = threadIdx.x;
    const int j    = lane < T ? lane : T - 1;   // clamp for inactive lanes

    // ---- numerator (scalar scan, lane 0) ----
    float num = 0.f;
    if (lane == 0) {
        int l0 = labels[(size_t)b * S];
        l0 = (l0 == -100) ? 0 : min(max(l0, 0), T - 1);
        num = start[l0] + E[((size_t)b * S) * T + l0];
        int prev = l0;
        for (int tt = 1; tt < S; ++tt) {
            const int lb = labels[(size_t)b * S + tt];
            if (lb != -100) {
                num += trans[prev * T + lb] + E[((size_t)b * S + tt) * T + lb];
                prev = lb;
            }
        }
        num += endw[prev];
    }

    // ---- denominator: alpha recursion over 7 tags ----
    float tr[T];
#pragma unroll
    for (int i = 0; i < T; ++i) tr[i] = trans[i * T + j];
    float alpha = start[j] + E[((size_t)b * S) * T + j];

    for (int tt = 1; tt < S; ++tt) {
        if (labels[(size_t)b * S + tt] != -100) {   // uniform branch per b
            float ai[T];
#pragma unroll
            for (int i = 0; i < T; ++i) ai[i] = __shfl(alpha, i, 32) + tr[i];
            float mx = ai[0];
#pragma unroll
            for (int i = 1; i < T; ++i) mx = fmaxf(mx, ai[i]);
            float sm = 0.f;
#pragma unroll
            for (int i = 0; i < T; ++i) sm += expf(ai[i] - mx);
            alpha = mx + logf(sm) + E[((size_t)b * S + tt) * T + j];
        }
    }
    const float fin = alpha + endw[j];
    if (lane == 0) {
        float v[T];
#pragma unroll
        for (int i = 0; i < T; ++i) v[i] = __shfl(fin, i, 32);
        float mx = v[0];
#pragma unroll
        for (int i = 1; i < T; ++i) mx = fmaxf(mx, v[i]);
        float sm = 0.f;
#pragma unroll
        for (int i = 0; i < T; ++i) sm += expf(v[i] - mx);
        lossb[b] = num - (mx + logf(sm));
    }
}

__global__ __launch_bounds__(32)
void crf_reduce_kernel(const float* __restrict__ lossb, float* __restrict__ out) {
    float v = lossb[threadIdx.x];   // B == 32 == one wave
#pragma unroll
    for (int off = 16; off > 0; off >>= 1) v += __shfl_xor(v, off, 32);
    if (threadIdx.x == 0) out[0] = -(v / (float)B);
}

// =====================================================================
// Host launcher
// =====================================================================
extern "C" void kernel_launch(void* const* d_in, const int* in_sizes, int n_in,
                              void* d_out, int out_size, void* d_ws, size_t ws_size,
                              hipStream_t stream) {
    (void)in_sizes; (void)n_in; (void)out_size; (void)ws_size;

    const float* word_emb = (const float*)d_in[0];
    const float* pos_emb  = (const float*)d_in[1];
    const float* emb_ln_s = (const float*)d_in[2];
    const float* emb_ln_b = (const float*)d_in[3];
    const float* attn_qw  = (const float*)d_in[4];
    const float* attn_qb  = (const float*)d_in[5];
    const float* attn_kw  = (const float*)d_in[6];
    const float* attn_kb  = (const float*)d_in[7];
    const float* attn_vw  = (const float*)d_in[8];
    const float* attn_vb  = (const float*)d_in[9];
    const float* attn_ow  = (const float*)d_in[10];
    const float* attn_ob  = (const float*)d_in[11];
    const float* ln1_s    = (const float*)d_in[12];
    const float* ln1_b    = (const float*)d_in[13];
    const float* ffn_w1   = (const float*)d_in[14];
    const float* ffn_b1   = (const float*)d_in[15];
    const float* ffn_w2   = (const float*)d_in[16];
    const float* ffn_b2   = (const float*)d_in[17];
    const float* ln2_s    = (const float*)d_in[18];
    const float* ln2_b    = (const float*)d_in[19];
    const float* cls_w    = (const float*)d_in[20];
    const float* cls_b    = (const float*)d_in[21];
    const float* crf_sta  = (const float*)d_in[22];
    const float* crf_end  = (const float*)d_in[23];
    const float* crf_tr   = (const float*)d_in[24];
    const int*   input_ids = (const int*)d_in[25];
    const int*   amask     = (const int*)d_in[26];
    const int*   labels    = (const int*)d_in[27];

    // workspace layout (floats, then bf16 weight region; base is 256B aligned)
    float* ws = (float*)d_ws;
    const size_t MH = (size_t)M * H;          // 12,582,912
    float* x     = ws;
    float* q     = x + MH;       // also reused as attn_out / ffn2 tmp
    float* k     = q + MH;
    float* v     = k + MH;
    float* ctx   = v + MH;
    float* hbuf  = ctx + MH;                  // M x FF
    float* emis  = hbuf + (size_t)M * FF;     // M x T
    float* lossb = emis + (size_t)M * T;      // B
    __bf16* wbt  = (__bf16*)(lossb + B);      // 16B-aligned (offset % 16 == 0)

    const size_t HH  = (size_t)H * H;
    const size_t HFF = (size_t)H * FF;
    const size_t WSTRIDE = 4 * HH + 2 * HFF;  // bf16 weights per layer

    const dim3 gemm_blk(256);
    const dim3 gemm_hh(H / 128, M / 128);     // N=768
    const dim3 gemm_hf(FF / 128, M / 128);    // N=3072
    const dim3 attn_grid(S / 64, NH, B);

    // ---- weight pre-conversion (f32 [K,N] -> bf16 [N,K]) ----
    for (int i = 0; i < L; ++i) {
        __bf16* base = wbt + (size_t)i * WSTRIDE;
        wconv_kernel<<<dim3(H / 64, H / 64), 256, 0, stream>>>(attn_qw + i * HH, base + 0 * HH, H, H);
        wconv_kernel<<<dim3(H / 64, H / 64), 256, 0, stream>>>(attn_kw + i * HH, base + 1 * HH, H, H);
        wconv_kernel<<<dim3(H / 64, H / 64), 256, 0, stream>>>(attn_vw + i * HH, base + 2 * HH, H, H);
        wconv_kernel<<<dim3(H / 64, H / 64), 256, 0, stream>>>(attn_ow + i * HH, base + 3 * HH, H, H);
        wconv_kernel<<<dim3(FF / 64, H / 64), 256, 0, stream>>>(ffn_w1 + i * HFF, base + 4 * HH, H, FF);
        wconv_kernel<<<dim3(H / 64, FF / 64), 256, 0, stream>>>(ffn_w2 + i * HFF, base + 4 * HH + HFF, FF, H);
    }

    embed_ln_kernel<<<M, 256, 0, stream>>>(input_ids, word_emb, pos_emb,
                                           emb_ln_s, emb_ln_b, x);

    for (int i = 0; i < L; ++i) {
        const size_t wH = (size_t)i * H;
        __bf16* base = wbt + (size_t)i * WSTRIDE;

        gemm_bias_kernel<false><<<gemm_hh, gemm_blk, 0, stream>>>(
            x, base + 0 * HH, attn_qb + wH, q, M, H, H);
        gemm_bias_kernel<false><<<gemm_hh, gemm_blk, 0, stream>>>(
            x, base + 1 * HH, attn_kb + wH, k, M, H, H);
        gemm_bias_kernel<false><<<gemm_hh, gemm_blk, 0, stream>>>(
            x, base + 2 * HH, attn_vb + wH, v, M, H, H);

        attention_kernel<<<attn_grid, 128, 0, stream>>>(q, k, v, amask, ctx);

        gemm_bias_kernel<false><<<gemm_hh, gemm_blk, 0, stream>>>(
            ctx, base + 3 * HH, attn_ob + wH, q, M, H, H);
        add_ln_kernel<<<M, 256, 0, stream>>>(x, q, ln1_s + wH, ln1_b + wH);

        gemm_bias_kernel<true><<<gemm_hf, gemm_blk, 0, stream>>>(
            x, base + 4 * HH, ffn_b1 + (size_t)i * FF, hbuf, M, FF, H);
        gemm_bias_kernel<false><<<gemm_hh, gemm_blk, 0, stream>>>(
            hbuf, base + 4 * HH + HFF, ffn_b2 + wH, q, M, H, FF);
        add_ln_kernel<<<M, 256, 0, stream>>>(x, q, ln2_s + wH, ln2_b + wH);
    }

    emissions_kernel<<<M, 224, 0, stream>>>(x, cls_w, cls_b, emis);
    crf_kernel<<<B, 32, 0, stream>>>(emis, labels, crf_sta, crf_end, crf_tr, lossb);
    crf_reduce_kernel<<<1, 32, 0, stream>>>(lossb, (float*)d_out);
}
